// TransformerEncoder_85710367359124
// MI455X (gfx1250) — matrix operand order repaired and checked by reading, not compiled, not run on hardware
//
#include <hip/hip_runtime.h>
#include <math.h>

// ---------------- problem constants ----------------
#define BB     8
#define NN     256
#define DD     384
#define LL     6
#define HH     8
#define FF_    1536
#define TT     4
#define NM1    517           // T + 1 + 2*N
#define S_TOT  1034          // 2*NM1
#define SPAD   1056          // S padded to multiple of 32 (33 chunks of 32)
#define MROWS  (BB * S_TOT)  // 8272, divisible by 16

typedef __attribute__((ext_vector_type(16))) __bf16 v16bf;
typedef __attribute__((ext_vector_type(8)))  __bf16 v8bf;
typedef __attribute__((ext_vector_type(8)))  float  v8f;

#define DEV static __device__ __forceinline__

DEV v16bf join16(v8bf lo, v8bf hi) {
    v16bf r;
#pragma unroll
    for (int i = 0; i < 8; ++i) { r[i] = lo[i]; r[i + 8] = hi[i]; }
    return r;
}

// A-matrix fragment (16x32 bf16): lane m = lane&15 holds row m.
// lanes 0-15: halves 0..7 = K 0..7, halves 8..15 = K 16..23
// lanes 16-31: halves 0..7 = K 8..15, halves 8..15 = K 24..31
DEV v16bf load_a_frag(const __bf16* rowbase, int lane) {
    int koff = (lane < 16) ? 0 : 8;
    v8bf lo = *(const v8bf*)(rowbase + koff);
    v8bf hi = *(const v8bf*)(rowbase + 16 + koff);
    return join16(lo, hi);
}

// B-matrix fragment (32x16 bf16): lane n = lane&15 holds column n.
// lanes 0-15: halves 0..7 = K 0..7, halves 8..15 = K 16..23
// lanes 16-31: halves 0..7 = K 8..15, halves 8..15 = K 24..31
// colbase points at (column row in O-major storage, element kbase); 16B aligned.
DEV v16bf load_b_frag(const __bf16* colbase, int lane) {
    int koff = (lane < 16) ? 0 : 8;
    v8bf lo = *(const v8bf*)(colbase + koff);
    v8bf hi = *(const v8bf*)(colbase + 16 + koff);
    return join16(lo, hi);
}

DEV v8f wmma_bf16(v16bf a, v16bf b, v8f c) {
    return __builtin_amdgcn_wmma_f32_16x16x32_bf16(false, a, false, b, (short)0, c,
                                                   false, false);
}

// ---------------- elementwise helpers ----------------
__global__ void cvt_bf16_kernel(const float* __restrict__ in, __bf16* __restrict__ out, int n) {
    int i = blockIdx.x * 256 + threadIdx.x;
    if (i < n) out[i] = (__bf16)in[i];
}

// Concatenate the 8 segments into src (B,S,D) and coords (B,S,3)
__global__ void build_src_kernel(
    const float* __restrict__ hand_t,  const float* __restrict__ head_t,
    const float* __restrict__ hand_m1, const float* __restrict__ head_m1,
    const float* __restrict__ c_hand_t,  const float* __restrict__ c_head_t,
    const float* __restrict__ c_hand_m1, const float* __restrict__ c_head_m1,
    const float* __restrict__ state_t, const float* __restrict__ state_m1,
    const float* __restrict__ trans_t, const float* __restrict__ trans_m1,
    const float* __restrict__ tok_m1,  const float* __restrict__ tok_t,
    float* __restrict__ src, float* __restrict__ coords)
{
    int s = blockIdx.x, b = blockIdx.y, d = threadIdx.x;
    const float* srow;
    const float* crow;
    if (s == 0)          { srow = state_m1 + (size_t)b * DD;                       crow = trans_m1 + b * 3; }
    else if (s <= 256)   { srow = hand_m1 + ((size_t)b * NN + (s - 1)) * DD;       crow = c_hand_m1 + ((size_t)b * NN + (s - 1)) * 3; }
    else if (s <= 512)   { srow = head_m1 + ((size_t)b * NN + (s - 257)) * DD;     crow = c_head_m1 + ((size_t)b * NN + (s - 257)) * 3; }
    else if (s <= 516)   { srow = tok_m1 + (size_t)(s - 513) * DD;                 crow = trans_m1 + b * 3; }
    else if (s == 517)   { srow = state_t + (size_t)b * DD;                        crow = trans_t + b * 3; }
    else if (s <= 773)   { srow = hand_t + ((size_t)b * NN + (s - 518)) * DD;      crow = c_hand_t + ((size_t)b * NN + (s - 518)) * 3; }
    else if (s <= 1029)  { srow = head_t + ((size_t)b * NN + (s - 774)) * DD;      crow = c_head_t + ((size_t)b * NN + (s - 774)) * 3; }
    else                 { srow = tok_t + (size_t)(s - 1030) * DD;                 crow = trans_t + b * 3; }
    src[((size_t)b * S_TOT + s) * DD + d] = srow[d];
    if (d < 3) coords[((size_t)b * S_TOT + s) * 3 + d] = crow[d];
}

// ---------------- GEMM: Y = A(bf16, MxK) * W^T(bf16, NoutxK) + bias [+resid] [gelu] --------
// EPI 0: f32 out = acc + bias
// EPI 1: f32 out = acc + bias + resid
// EPI 2: bf16 out = gelu(acc + bias)
template <int EPI>
__global__ __launch_bounds__(256) void gemm_bf16_kernel(
    const __bf16* __restrict__ A, const __bf16* __restrict__ W,
    const float* __restrict__ bias, const float* __restrict__ resid,
    float* __restrict__ outF, __bf16* __restrict__ outB,
    int M, int K, int Nout)
{
    int lane = threadIdx.x & 31;
    int wave = threadIdx.x >> 5;
    int row0 = blockIdx.y * 128 + (wave & 3) * 32;
    int col0 = blockIdx.x * 128 + (wave >> 2) * 64;
    if (row0 >= M || col0 >= Nout) return;
    bool m1ok = (row0 + 32) <= M;  // second 16-row tile fully valid (M % 16 == 0)

    v8f acc[2][4];
#pragma unroll
    for (int a = 0; a < 2; ++a)
#pragma unroll
        for (int b = 0; b < 4; ++b)
#pragma unroll
            for (int r = 0; r < 8; ++r) acc[a][b][r] = 0.0f;

    int mA = lane & 15;
    const __bf16* arow0 = A + (size_t)(row0 + mA) * K;
    const __bf16* arow1 = A + (size_t)(row0 + 16 + mA) * K;
    const __bf16* wrow[4];
#pragma unroll
    for (int nt = 0; nt < 4; ++nt) wrow[nt] = W + (size_t)(col0 + nt * 16 + mA) * K;

    for (int kb = 0; kb < K; kb += 32) {
        if (kb + 64 < K) __builtin_prefetch(arow0 + kb + 64, 0, 0);  // global_prefetch_b8
        v16bf a0 = load_a_frag(arow0 + kb, lane);
        v16bf a1 = m1ok ? load_a_frag(arow1 + kb, lane) : a0;
#pragma unroll
        for (int nt = 0; nt < 4; ++nt) {
            v16bf bfrag = load_b_frag(wrow[nt] + kb, lane);
            acc[0][nt] = wmma_bf16(a0, bfrag, acc[0][nt]);
            acc[1][nt] = wmma_bf16(a1, bfrag, acc[1][nt]);
        }
    }

    int hi = lane >> 4;
    int ncol = lane & 15;
#pragma unroll
    for (int mt = 0; mt < 2; ++mt) {
        if (mt == 1 && !m1ok) break;
#pragma unroll
        for (int nt = 0; nt < 4; ++nt) {
            int col = col0 + nt * 16 + ncol;
            float bv = bias[col];
#pragma unroll
            for (int r = 0; r < 8; ++r) {
                int row = row0 + mt * 16 + hi * 8 + r;
                float v = acc[mt][nt][r] + bv;
                if (EPI == 2) {
                    float g = 0.5f * v * (1.0f + erff(v * 0.70710678118654752f));
                    outB[(size_t)row * Nout + col] = (__bf16)g;
                } else {
                    if (EPI == 1) v += resid[(size_t)row * Nout + col];
                    outF[(size_t)row * Nout + col] = v;
                }
            }
        }
    }
}

// ---------------- rotary-3D + pack to WMMA-friendly layouts ----------------
// qp/kp: [B*H][SPAD][64] bf16 (Dh=48 zero-padded to 64), vt: [B*H][48][SPAD] bf16
__global__ __launch_bounds__(64) void rotary_pack_kernel(
    const float* __restrict__ qf, const float* __restrict__ kf,
    const float* __restrict__ vf, const float* __restrict__ coords,
    __bf16* __restrict__ qp, __bf16* __restrict__ kp, __bf16* __restrict__ vt)
{
    int idx = blockIdx.x * 64 + threadIdx.x;  // over B*H*SPAD
    if (idx >= BB * HH * SPAD) return;
    int s = idx % SPAD;
    int bh = idx / SPAD;
    int h = bh % HH, b = bh / HH;
    __bf16* qrow = qp + (size_t)idx * 64;
    __bf16* krow = kp + (size_t)idx * 64;
    __bf16* vcol = vt + (size_t)bh * 48 * SPAD + s;
    if (s >= S_TOT) {
#pragma unroll
        for (int d = 0; d < 64; ++d) { qrow[d] = (__bf16)0.0f; krow[d] = (__bf16)0.0f; }
        for (int d = 0; d < 48; ++d) vcol[(size_t)d * SPAD] = (__bf16)0.0f;
        return;
    }
    const float* qs = qf + ((size_t)b * S_TOT + s) * DD + h * 48;
    const float* ks = kf + ((size_t)b * S_TOT + s) * DD + h * 48;
    const float* vs = vf + ((size_t)b * S_TOT + s) * DD + h * 48;
    const float* cc = coords + ((size_t)b * S_TOT + s) * 3;
    float qo[48], ko[48];
#pragma unroll
    for (int a = 0; a < 3; ++a) {
        float ca = cc[a];
#pragma unroll
        for (int i = 0; i < 8; ++i) {
            float invf = __powf(10000.0f, -(float)i / 8.0f);
            float ang = ca * invf;
            float cs = __cosf(ang), sn = __sinf(ang);
            float q1 = qs[a * 16 + i], q2 = qs[a * 16 + 8 + i];
            float k1 = ks[a * 16 + i], k2 = ks[a * 16 + 8 + i];
            qo[a * 16 + i]     = q1 * cs - q2 * sn;
            qo[a * 16 + 8 + i] = q1 * sn + q2 * cs;
            ko[a * 16 + i]     = k1 * cs - k2 * sn;
            ko[a * 16 + 8 + i] = k1 * sn + k2 * cs;
        }
    }
#pragma unroll
    for (int d = 0; d < 48; ++d) {
        qrow[d] = (__bf16)qo[d];
        krow[d] = (__bf16)ko[d];
        vcol[(size_t)d * SPAD] = (__bf16)vs[d];
    }
#pragma unroll
    for (int d = 48; d < 64; ++d) { qrow[d] = (__bf16)0.0f; krow[d] = (__bf16)0.0f; }
}

// ---------------- flash attention, one wave per (b, h, 16-query tile) ----------------
__global__ __launch_bounds__(32) void attn_kernel(
    const __bf16* __restrict__ qp, const __bf16* __restrict__ kp,
    const __bf16* __restrict__ vt, __bf16* __restrict__ attnB)
{
    __shared__ __attribute__((aligned(16))) __bf16 pb[16 * 32];  // P staging tile
    int lane = threadIdx.x & 31;
    int qt = blockIdx.x, h = blockIdx.y, b = blockIdx.z;
    int mA = lane & 15, hi = lane >> 4;

    const __bf16* qbase = qp + ((size_t)(b * HH + h) * SPAD + qt * 16) * 64;
    const __bf16* kbase = kp + (size_t)(b * HH + h) * SPAD * 64;
    const __bf16* vbase = vt + (size_t)(b * HH + h) * 48 * SPAD;

    v16bf qa0 = load_a_frag(qbase + (size_t)mA * 64 + 0, lane);
    v16bf qa1 = load_a_frag(qbase + (size_t)mA * 64 + 32, lane);

    float Mrow[8], Lrow[8];
    v8f Oacc[3];
#pragma unroll
    for (int r = 0; r < 8; ++r) { Mrow[r] = -1e30f; Lrow[r] = 0.0f; }
#pragma unroll
    for (int nt = 0; nt < 3; ++nt)
#pragma unroll
        for (int r = 0; r < 8; ++r) Oacc[nt][r] = 0.0f;

    const float rs = 0.14433756729740643f;  // 1/sqrt(48)
    int rowbase = qt * 16 + hi * 8;

    for (int c = 0; c < SPAD / 32; ++c) {
        int kv0 = c * 32;
        v8f s[2];
#pragma unroll
        for (int t = 0; t < 2; ++t) {
            const __bf16* kr = kbase + (size_t)(kv0 + t * 16 + mA) * 64;
            v16bf kb0 = load_b_frag(kr + 0, lane);
            v16bf kb1 = load_b_frag(kr + 32, lane);
            v8f z;
#pragma unroll
            for (int r = 0; r < 8; ++r) z[r] = 0.0f;
            z = wmma_bf16(qa0, kb0, z);
            s[t] = wmma_bf16(qa1, kb1, z);
        }
        int col0 = kv0 + (lane & 15);
        int col1 = col0 + 16;
#pragma unroll
        for (int r = 0; r < 8; ++r) {
            int row = rowbase + r;
            float v0 = s[0][r] * rs, v1 = s[1][r] * rs;
            bool qm = row < NM1;
            if (col0 >= S_TOT || (qm && col0 >= NM1)) v0 = -1e30f;
            if (col1 >= S_TOT || (qm && col1 >= NM1)) v1 = -1e30f;
            float tmax = fmaxf(v0, v1);
#pragma unroll
            for (int msk = 8; msk >= 1; msk >>= 1) tmax = fmaxf(tmax, __shfl_xor(tmax, msk, 16));
            float mn = fmaxf(Mrow[r], tmax);
            float sc = __expf(Mrow[r] - mn);
            float e0 = __expf(v0 - mn);
            float e1 = __expf(v1 - mn);
            float su = e0 + e1;
#pragma unroll
            for (int msk = 8; msk >= 1; msk >>= 1) su += __shfl_xor(su, msk, 16);
            Lrow[r] = Lrow[r] * sc + su;
            Mrow[r] = mn;
#pragma unroll
            for (int nt = 0; nt < 3; ++nt) Oacc[nt][r] *= sc;
            pb[(r + hi * 8) * 32 + (lane & 15)] = (__bf16)e0;
            pb[(r + hi * 8) * 32 + 16 + (lane & 15)] = (__bf16)e1;
        }
        __syncthreads();
        {
            int koff = (lane < 16) ? 0 : 8;
            const __bf16* pr = pb + mA * 32;
            v8bf lo = *(const v8bf*)(pr + koff);
            v8bf hh = *(const v8bf*)(pr + 16 + koff);
            v16bf pa = join16(lo, hh);
#pragma unroll
            for (int nt = 0; nt < 3; ++nt) {
                const __bf16* vr = vbase + (size_t)(nt * 16 + mA) * SPAD + kv0;
                v16bf vb = load_b_frag(vr, lane);
                Oacc[nt] = wmma_bf16(pa, vb, Oacc[nt]);
            }
        }
        __syncthreads();
    }

#pragma unroll
    for (int nt = 0; nt < 3; ++nt) {
        int col = h * 48 + nt * 16 + (lane & 15);
#pragma unroll
        for (int r = 0; r < 8; ++r) {
            int row = qt * 16 + hi * 8 + r;
            if (row < S_TOT) {
                float v = Oacc[nt][r] / Lrow[r];
                attnB[((size_t)b * S_TOT + row) * DD + col] = (__bf16)v;
            }
        }
    }
}

// ---------------- LayerNorm (dual output: f32 + bf16 for next GEMM) ----------------
__global__ __launch_bounds__(128) void layernorm_dual_kernel(
    const float* __restrict__ in, const float* __restrict__ g, const float* __restrict__ be,
    float* __restrict__ outF, __bf16* __restrict__ outB, int M)
{
    int lane = threadIdx.x & 31;
    int wave = threadIdx.x >> 5;
    int row = blockIdx.x * 4 + wave;
    if (row >= M) return;
    const float* x = in + (size_t)row * DD;
    float s = 0.0f, s2 = 0.0f;
    for (int i = lane; i < DD; i += 32) { float v = x[i]; s += v; s2 += v * v; }
#pragma unroll
    for (int msk = 16; msk >= 1; msk >>= 1) { s += __shfl_xor(s, msk, 32); s2 += __shfl_xor(s2, msk, 32); }
    float mean = s * (1.0f / DD);
    float var = s2 * (1.0f / DD) - mean * mean;
    float inv = rsqrtf(var + 1e-5f);
    for (int i = lane; i < DD; i += 32) {
        float v = (x[i] - mean) * inv * g[i] + be[i];
        outF[(size_t)row * DD + i] = v;
        outB[(size_t)row * DD + i] = (__bf16)v;
    }
}

__global__ void copy_out_kernel(const float* __restrict__ src, float* __restrict__ out) {
    int t = blockIdx.x, b = blockIdx.y, d = threadIdx.x;
    out[((size_t)b * TT + t) * DD + d] = src[((size_t)b * S_TOT + (S_TOT - TT) + t) * DD + d];
}

// ---------------- host-side orchestration ----------------
extern "C" void kernel_launch(void* const* d_in, const int* in_sizes, int n_in,
                              void* d_out, int out_size, void* d_ws, size_t ws_size,
                              hipStream_t stream) {
    (void)in_sizes; (void)n_in; (void)out_size; (void)ws_size;
    const float* hand_t   = (const float*)d_in[0];
    const float* head_t   = (const float*)d_in[1];
    const float* hand_m1  = (const float*)d_in[2];
    const float* head_m1  = (const float*)d_in[3];
    const float* c_hand_t = (const float*)d_in[4];
    const float* c_head_t = (const float*)d_in[5];
    const float* c_hand_m1= (const float*)d_in[6];
    const float* c_head_m1= (const float*)d_in[7];
    const float* state_t  = (const float*)d_in[8];
    const float* state_m1 = (const float*)d_in[9];
    const float* trans_t  = (const float*)d_in[10];
    const float* trans_m1 = (const float*)d_in[11];
    const float* tok_m1   = (const float*)d_in[12];
    const float* tok_t    = (const float*)d_in[13];
    const float* Wq = (const float*)d_in[14]; const float* bq = (const float*)d_in[15];
    const float* Wk = (const float*)d_in[16]; const float* bk = (const float*)d_in[17];
    const float* Wv = (const float*)d_in[18]; const float* bv = (const float*)d_in[19];
    const float* Wo = (const float*)d_in[20]; const float* bo = (const float*)d_in[21];
    const float* W1 = (const float*)d_in[22]; const float* b1 = (const float*)d_in[23];
    const float* W2 = (const float*)d_in[24]; const float* b2 = (const float*)d_in[25];
    const float* g1 = (const float*)d_in[26]; const float* be1 = (const float*)d_in[27];
    const float* g2 = (const float*)d_in[28]; const float* be2 = (const float*)d_in[29];

    // carve workspace
    char* base = (char*)d_ws;
    size_t off = 0;
    auto alloc = [&](size_t bytes) -> char* {
        char* p = base + off;
        off = (off + bytes + 255) & ~(size_t)255;
        return p;
    };
    const size_t MD = (size_t)MROWS * DD;
    float*  src_f  = (float*)alloc(MD * 4);
    float*  coords = (float*)alloc((size_t)MROWS * 3 * 4);
    __bf16* x_b    = (__bf16*)alloc(MD * 2);
    float*  qf     = (float*)alloc(MD * 4);   // later reused as res1
    float*  kf     = (float*)alloc(MD * 4);   // later reused as src2_f
    float*  vf     = (float*)alloc(MD * 4);   // later reused as res2
    __bf16* qp     = (__bf16*)alloc((size_t)BB * HH * SPAD * 64 * 2);
    __bf16* kp     = (__bf16*)alloc((size_t)BB * HH * SPAD * 64 * 2);
    __bf16* vt     = (__bf16*)alloc((size_t)BB * HH * 48 * SPAD * 2);
    __bf16* attn_b = (__bf16*)alloc(MD * 2);
    __bf16* src2_b = (__bf16*)alloc(MD * 2);
    __bf16* ff1_b  = (__bf16*)alloc((size_t)MROWS * FF_ * 2);
    __bf16* Wq_b = (__bf16*)alloc((size_t)LL * DD * DD * 2);
    __bf16* Wk_b = (__bf16*)alloc((size_t)LL * DD * DD * 2);
    __bf16* Wv_b = (__bf16*)alloc((size_t)LL * DD * DD * 2);
    __bf16* Wo_b = (__bf16*)alloc((size_t)LL * DD * DD * 2);
    __bf16* W1_b = (__bf16*)alloc((size_t)LL * FF_ * DD * 2);
    __bf16* W2_b = (__bf16*)alloc((size_t)LL * DD * FF_ * 2);
    float* res1   = qf;
    float* src2_f = kf;
    float* res2   = vf;

    auto cvt = [&](const float* in, __bf16* out, size_t n) {
        cvt_bf16_kernel<<<dim3((unsigned)((n + 255) / 256)), dim3(256), 0, stream>>>(in, out, (int)n);
    };
    cvt(Wq, Wq_b, (size_t)LL * DD * DD);
    cvt(Wk, Wk_b, (size_t)LL * DD * DD);
    cvt(Wv, Wv_b, (size_t)LL * DD * DD);
    cvt(Wo, Wo_b, (size_t)LL * DD * DD);
    cvt(W1, W1_b, (size_t)LL * FF_ * DD);
    cvt(W2, W2_b, (size_t)LL * DD * FF_);

    build_src_kernel<<<dim3(S_TOT, BB), dim3(DD), 0, stream>>>(
        hand_t, head_t, hand_m1, head_m1, c_hand_t, c_head_t, c_hand_m1, c_head_m1,
        state_t, state_m1, trans_t, trans_m1, tok_m1, tok_t, src_f, coords);
    cvt(src_f, x_b, MD);

    dim3 gemmGridD(3, (MROWS + 127) / 128);    // Nout = 384
    dim3 gemmGridF(12, (MROWS + 127) / 128);   // Nout = 1536
    dim3 gemmBlock(256);

    for (int i = 0; i < LL; ++i) {
        const __bf16* wq = Wq_b + (size_t)i * DD * DD;
        const __bf16* wk = Wk_b + (size_t)i * DD * DD;
        const __bf16* wv = Wv_b + (size_t)i * DD * DD;
        const __bf16* wo = Wo_b + (size_t)i * DD * DD;
        const __bf16* w1 = W1_b + (size_t)i * FF_ * DD;
        const __bf16* w2 = W2_b + (size_t)i * DD * FF_;

        gemm_bf16_kernel<0><<<gemmGridD, gemmBlock, 0, stream>>>(
            x_b, wq, bq + i * DD, nullptr, qf, nullptr, MROWS, DD, DD);
        gemm_bf16_kernel<0><<<gemmGridD, gemmBlock, 0, stream>>>(
            x_b, wk, bk + i * DD, nullptr, kf, nullptr, MROWS, DD, DD);
        gemm_bf16_kernel<0><<<gemmGridD, gemmBlock, 0, stream>>>(
            x_b, wv, bv + i * DD, nullptr, vf, nullptr, MROWS, DD, DD);

        rotary_pack_kernel<<<dim3((BB * HH * SPAD + 63) / 64), dim3(64), 0, stream>>>(
            qf, kf, vf, coords, qp, kp, vt);

        attn_kernel<<<dim3((S_TOT + 15) / 16, HH, BB), dim3(32), 0, stream>>>(
            qp, kp, vt, attn_b);

        gemm_bf16_kernel<1><<<gemmGridD, gemmBlock, 0, stream>>>(
            attn_b, wo, bo + i * DD, src_f, res1, nullptr, MROWS, DD, DD);

        layernorm_dual_kernel<<<dim3((MROWS + 3) / 4), dim3(128), 0, stream>>>(
            res1, g1 + i * DD, be1 + i * DD, src2_f, src2_b, MROWS);

        gemm_bf16_kernel<2><<<gemmGridF, gemmBlock, 0, stream>>>(
            src2_b, w1, b1 + i * FF_, nullptr, nullptr, ff1_b, MROWS, DD, FF_);

        gemm_bf16_kernel<1><<<gemmGridD, gemmBlock, 0, stream>>>(
            ff1_b, w2, b2 + i * DD, src2_f, res2, nullptr, MROWS, FF_, DD);

        layernorm_dual_kernel<<<dim3((MROWS + 3) / 4), dim3(128), 0, stream>>>(
            res2, g2 + i * DD, be2 + i * DD, src_f, x_b, MROWS);
    }

    copy_out_kernel<<<dim3(TT, BB), dim3(DD), 0, stream>>>(src_f, (float*)d_out);
}
